// GrokDecoderLayer_30674656428589
// MI455X (gfx1250) — matrix operand order repaired and checked by reading, not compiled, not run on hardware
//
#include <hip/hip_runtime.h>
#include <hip/hip_bf16.h>
#include <cstdint>

// ---------------- problem constants (from reference) ----------------
constexpr int Md  = 1024;          // model dim M
constexpr int Gc  = 8;             // groups
constexpr int Ec  = 8;             // experts
constexpr int Sc  = 1024;          // tokens per group (B*L/G)
constexpr int Cc  = 256;           // capacity  (int(2.0*S/E) rounded to mult 4)
constexpr int Hc  = 5464;          // ffn hidden
constexpr int HP  = 5504;          // H padded to multiple of 128 (43*128)
constexpr int GSc = Gc * Sc;       // 8192 tokens
constexpr int RPE = Gc * Cc;       // 2048 rows per expert
constexpr int PITCH = 40;          // LDS row pitch in bf16 elems (80B -> bank stride 20)

typedef __attribute__((ext_vector_type(16))) __bf16 v16bf;
typedef __attribute__((ext_vector_type(8)))  __bf16 v8bf;
typedef __attribute__((ext_vector_type(8)))  float  v8f;

// ---------------- CDNA5 async global->LDS copy (ASYNCcnt tracked) ------------
__device__ __forceinline__ void async_ld_b128(void* lds, const void* gptr) {
  const uint32_t l = (uint32_t)(uintptr_t)lds;       // LDS byte offset
  const uint64_t g = (uint64_t)(uintptr_t)gptr;      // global address
  asm volatile("global_load_async_to_lds_b128 %0, %1, off"
               :: "v"(l), "v"(g) : "memory");
}
__device__ __forceinline__ void wait_async0() {
  asm volatile("s_wait_asynccnt 0x0" ::: "memory");
}

// ---------------- WMMA helpers (CDNA5 16x16x32 bf16, fp32 acc) ----------------
__device__ __forceinline__ v16bf cat16(v8bf lo, v8bf hi) {
  v16bf r;
#pragma unroll
  for (int i = 0; i < 8; ++i) { r[i] = lo[i]; r[i + 8] = hi[i]; }
  return r;
}

__device__ __forceinline__ v8f wmma_bf16(v16bf a, v16bf b, v8f c) {
  return __builtin_amdgcn_wmma_f32_16x16x32_bf16(false, a, false, b, (short)0, c,
                                                 false, false);
}

// A fragment from an LDS tile: row-major [128][PITCH], 16(M)x32(K) subtile.
// lane l: m = l&15 ; kb = (l>>4)*8 ; holds K = kb..kb+7 and kb+16..kb+23.
__device__ __forceinline__ v16bf a_frag_lds(const __bf16* tile, int row0, int lane) {
  const int m  = lane & 15;
  const int kb = (lane >> 4) << 3;
  const __bf16* p = tile + (size_t)(row0 + m) * PITCH + kb;
  return cat16(*(const v8bf*)p, *(const v8bf*)(p + 16));
}
// B fragment from an LDS tile holding a (N x K) slab: 32(K)x16(N) subtile.
// lane l: n = l&15 ; kb = (l>>4)*8 ; holds K = kb..kb+7 and kb+16..kb+23.
__device__ __forceinline__ v16bf b_frag_lds(const __bf16* tile, int nrow0, int lane) {
  const int n  = lane & 15;
  const int kb = (lane >> 4) << 3;
  const __bf16* p = tile + (size_t)(nrow0 + n) * PITCH + kb;
  return cat16(*(const v8bf*)p, *(const v8bf*)(p + 16));
}

__device__ __forceinline__ float gelu_tanh(float x) {
  const float x3 = x * x * x;
  return 0.5f * x * (1.0f + tanhf(0.7978845608028654f * (x + 0.044715f * x3)));
}

// ---------------- 1) router logits ----------------
__global__ __launch_bounds__(256) void router_kernel(
    const float* __restrict__ x, const float* __restrict__ rw,
    float* __restrict__ logits) {
  const int gid = blockIdx.x * 256 + threadIdx.x;   // over G*S*E
  const int e  = gid & 7;
  const int gs = gid >> 3;
  const float* xr = x + (size_t)gs * Md;
  float acc = 0.f;
#pragma unroll 4
  for (int m = 0; m < Md; ++m) acc = fmaf(xr[m], rw[m * Ec + e], acc);
  logits[gid] = acc;
}

// ---------------- 2) top-2 gating with capacity (one block per group) ----------
__global__ __launch_bounds__(256) void gating_kernel(
    const float* __restrict__ logits, int* __restrict__ slot,
    int* __restrict__ tokE, int* __restrict__ tokC, float* __restrict__ tokG) {
  const int g = blockIdx.x;
  const int t = threadIdx.x;                       // owns tokens 4t..4t+3
  __shared__ unsigned sc[256];
  __shared__ unsigned tot1s[8];

  for (int i = t; i < Ec * Cc; i += 256) {
    const int e = i / Cc, c = i % Cc;
    slot[((size_t)e * Gc + g) * Cc + c] = -1;
  }

  int   idx1[4], idx2[4];
  float gv1[4],  gv2[4];
#pragma unroll
  for (int i = 0; i < 4; ++i) {
    const int s = t * 4 + i;
    const float* lp = logits + ((size_t)g * Sc + s) * Ec;
    float p[8];
    float mx = lp[0];
#pragma unroll
    for (int e = 0; e < 8; ++e) { p[e] = lp[e]; mx = fmaxf(mx, p[e]); }
    float sum = 0.f;
#pragma unroll
    for (int e = 0; e < 8; ++e) { p[e] = __expf(p[e] - mx); sum += p[e]; }
    const float inv = 1.f / sum;
#pragma unroll
    for (int e = 0; e < 8; ++e) p[e] *= inv;
    int i1 = 0; float b1 = p[0];
#pragma unroll
    for (int e = 1; e < 8; ++e) if (p[e] > b1) { b1 = p[e]; i1 = e; }
    int i2 = 0; float b2 = -1.f;
#pragma unroll
    for (int e = 0; e < 8; ++e)
      if (e != i1 && p[e] > b2) { b2 = p[e]; i2 = e; }
    idx1[i] = i1; gv1[i] = b1;
    idx2[i] = i2; gv2[i] = b2;
  }

  int pos1[4], pos2[4];
  for (int e = 0; e < 8; ++e) {
    unsigned runv[4]; unsigned mycnt = 0;
#pragma unroll
    for (int i = 0; i < 4; ++i) { runv[i] = mycnt; mycnt += (idx1[i] == e); }
    sc[t] = mycnt; __syncthreads();
    for (int off = 1; off < 256; off <<= 1) {
      unsigned v = (t >= off) ? sc[t - off] : 0u;
      __syncthreads();
      sc[t] += v;
      __syncthreads();
    }
    const unsigned excl = sc[t] - mycnt;
    if (t == 255) tot1s[e] = sc[255];
#pragma unroll
    for (int i = 0; i < 4; ++i) if (idx1[i] == e) pos1[i] = (int)(excl + runv[i]);
    __syncthreads();
  }
  for (int e = 0; e < 8; ++e) {
    unsigned runv[4]; unsigned mycnt = 0;
#pragma unroll
    for (int i = 0; i < 4; ++i) { runv[i] = mycnt; mycnt += (idx2[i] == e); }
    sc[t] = mycnt; __syncthreads();
    for (int off = 1; off < 256; off <<= 1) {
      unsigned v = (t >= off) ? sc[t - off] : 0u;
      __syncthreads();
      sc[t] += v;
      __syncthreads();
    }
    const unsigned excl = sc[t] - mycnt;
#pragma unroll
    for (int i = 0; i < 4; ++i) if (idx2[i] == e) pos2[i] = (int)(excl + runv[i]);
    __syncthreads();
  }

#pragma unroll
  for (int i = 0; i < 4; ++i) {
    const int s  = t * 4 + i;
    const int sg = g * Sc + s;
    const int e1 = idx1[i];
    const int p1 = pos1[i];
    const bool k1 = (p1 < Cc);
    tokE[sg] = k1 ? e1 : -1;
    tokC[sg] = p1;
    tokG[sg] = k1 ? gv1[i] : 0.f;
    if (k1) slot[((size_t)e1 * Gc + g) * Cc + p1] = s;

    const int e2  = idx2[i];
    const int m1c = min((int)tot1s[e2], Cc);    // capped top-1 count
    const int p2  = pos2[i] + m1c;
    const bool k2 = (p2 < Cc);
    tokE[GSc + sg] = k2 ? e2 : -1;
    tokC[GSc + sg] = p2;
    tokG[GSc + sg] = k2 ? gv2[i] : 0.f;
    if (k2) slot[((size_t)e2 * Gc + g) * Cc + p2] = s;
  }
}

// ---------------- 3) gather tokens into expert rows (fp32 -> bf16) -----------
__global__ __launch_bounds__(128) void gather_kernel(
    const float* __restrict__ x, const int* __restrict__ slot,
    __bf16* __restrict__ Xb) {
  const int row = blockIdx.x;              // row = e*(G*C) + g*C + c
  const int g   = (row / Cc) % Gc;
  const int s   = slot[row];
  __bf16* dst = Xb + (size_t)row * Md + threadIdx.x * 8;
  if (s < 0) {
    v8bf z = {};
    *(v8bf*)dst = z;
  } else {
    const float* src = x + ((size_t)g * Sc + s) * Md + threadIdx.x * 8;
    v8bf v;
#pragma unroll
    for (int i = 0; i < 8; ++i) v[i] = (__bf16)src[i];
    *(v8bf*)dst = v;
  }
}

// ---------------- 4) transpose + convert weights (fp32 -> bf16, zero pad) ----
// out[r][c] = (r < Cin && c < Rin) ? in[c][r] : 0   (per expert slab)
__global__ __launch_bounds__(256) void transpose_cvt_kernel(
    const float* __restrict__ in, __bf16* __restrict__ out,
    int Rin, int Cin, int ORows, int OCols) {
  __shared__ float tile[32][33];
  const int e = blockIdx.z;
  in  += (size_t)e * (size_t)Rin * Cin;
  out += (size_t)e * (size_t)ORows * OCols;
  const int c0 = blockIdx.x * 32;          // out col base
  const int r0 = blockIdx.y * 32;          // out row base
  const int tx = threadIdx.x & 31;
  const int ty = threadIdx.x >> 5;         // 0..7
#pragma unroll
  for (int k = 0; k < 4; ++k) {
    const int ir = c0 + ty + k * 8;        // input row index
    const int ic = r0 + tx;                // input col index
    float v = 0.f;
    if (ir < Rin && ic < Cin) v = in[(size_t)ir * Cin + ic];
    tile[ty + k * 8][tx] = v;
  }
  __syncthreads();
#pragma unroll
  for (int k = 0; k < 4; ++k) {
    const int orr = r0 + ty + k * 8;
    const int oc  = c0 + tx;
    if (orr < ORows && oc < OCols)
      out[(size_t)orr * OCols + oc] = (__bf16)tile[tx][ty + k * 8];
  }
}

// ---------------- 5) GEMM1: hidden = gelu(X@W0) * (X@W1)  (bf16 out) ---------
// 8 waves (4 along M, 2 along N) -> 128x128 block tile; wave tile 32x64.
// K sliced in 32-wide stages, double-buffered in LDS via async global->LDS.
// Fragments are preloaded into distinct registers so the WMMAs issue densely.
__global__ __launch_bounds__(256) void gemm1_kernel(
    const __bf16* __restrict__ Xb, const __bf16* __restrict__ W0T,
    const __bf16* __restrict__ W1T, __bf16* __restrict__ Hid) {
  constexpr int K = Md;                    // 1024
  constexpr int N = HP;                    // 5504
  __shared__ __bf16 lA [2][128 * PITCH];
  __shared__ __bf16 lB0[2][128 * PITCH];
  __shared__ __bf16 lB1[2][128 * PITCH];

  const int e = blockIdx.z;
  const __bf16* A  = Xb  + (size_t)e * RPE * K + (size_t)blockIdx.y * 128 * K;
  const __bf16* B0 = W0T + (size_t)e * (size_t)N * K + (size_t)blockIdx.x * 128 * K;
  const __bf16* B1 = W1T + (size_t)e * (size_t)N * K + (size_t)blockIdx.x * 128 * K;
  __bf16* Ho = Hid + (size_t)e * (size_t)RPE * N;

  const int tid  = threadIdx.x;
  const int lane = tid & 31;
  const int wid  = tid >> 5;
  const int wrow = (wid & 3) * 32;         // wave row inside block tile
  const int wcol = (wid >> 2) * 64;        // wave col inside block tile
  const int rq   = tid >> 2;               // staging chunk row (c=0)
  const int kq   = (tid & 3) * 8;          // staging chunk k-offset (elems)

  auto issue = [&](int k0, int buf) {
#pragma unroll
    for (int c = 0; c < 2; ++c) {
      const int r = rq + 64 * c;
      async_ld_b128(&lA [buf][r * PITCH + kq], A  + (size_t)r * K + k0 + kq);
      async_ld_b128(&lB0[buf][r * PITCH + kq], B0 + (size_t)r * K + k0 + kq);
      async_ld_b128(&lB1[buf][r * PITCH + kq], B1 + (size_t)r * K + k0 + kq);
    }
  };

  const v8f vzero = {0.f, 0.f, 0.f, 0.f, 0.f, 0.f, 0.f, 0.f};
  v8f acc0[2][4], acc1[2][4];
#pragma unroll
  for (int i = 0; i < 2; ++i)
#pragma unroll
    for (int j = 0; j < 4; ++j) { acc0[i][j] = vzero; acc1[i][j] = vzero; }

  issue(0, 0);
  wait_async0();
  __syncthreads();

  for (int k0 = 0; k0 < K; k0 += 32) {
    const int cur = (k0 >> 5) & 1;
    if (k0 + 32 < K) issue(k0 + 32, cur ^ 1);   // prefetch next stage into LDS

    // preload ALL fragments for this K-step (one ds clause, one wait)
    v16bf a0, a1, b0f[4], b1f[4];
    a0 = a_frag_lds(lA[cur], wrow,      lane);
    a1 = a_frag_lds(lA[cur], wrow + 16, lane);
#pragma unroll
    for (int j = 0; j < 4; ++j) {
      b0f[j] = b_frag_lds(lB0[cur], wcol + 16 * j, lane);
      b1f[j] = b_frag_lds(lB1[cur], wcol + 16 * j, lane);
    }
    // dense WMMA burst (C-chaining only; no D->A/B hazard)
#pragma unroll
    for (int j = 0; j < 4; ++j) {
      acc0[0][j] = wmma_bf16(a0, b0f[j], acc0[0][j]);
      acc0[1][j] = wmma_bf16(a1, b0f[j], acc0[1][j]);
      acc1[0][j] = wmma_bf16(a0, b1f[j], acc1[0][j]);
      acc1[1][j] = wmma_bf16(a1, b1f[j], acc1[1][j]);
    }
    wait_async0();
    __syncthreads();
  }

  const int row0 = blockIdx.y * 128 + wrow;
  const int col0 = blockIdx.x * 128 + wcol;
  const int n0 = lane & 15;
  const int ms = (lane >> 4) << 3;
#pragma unroll
  for (int i = 0; i < 2; ++i)
#pragma unroll
    for (int j = 0; j < 4; ++j)
#pragma unroll
      for (int tq = 0; tq < 8; ++tq) {
        const float h = gelu_tanh(acc0[i][j][tq]) * acc1[i][j][tq];
        const int r  = row0 + i * 16 + ms + tq;
        const int cc = col0 + j * 16 + n0;
        Ho[(size_t)r * N + cc] = (__bf16)h;
      }
}

// ---------------- 6) GEMM2: Y = hidden @ Wo  (fp32 out) ----------------------
// 4 waves (2x2) -> 128x128 block tile; wave tile 64x64 (16 WMMA : 16 ds_load).
__global__ __launch_bounds__(128) void gemm2_kernel(
    const __bf16* __restrict__ Hid, const __bf16* __restrict__ WoT,
    float* __restrict__ Y) {
  constexpr int K = HP;                    // 5504
  constexpr int N = Md;                    // 1024
  __shared__ __bf16 lA[2][128 * PITCH];
  __shared__ __bf16 lB[2][128 * PITCH];

  const int e = blockIdx.z;
  const __bf16* A  = Hid + (size_t)e * (size_t)RPE * K + (size_t)blockIdx.y * 128 * K;
  const __bf16* BT = WoT + (size_t)e * (size_t)N * K + (size_t)blockIdx.x * 128 * K;
  float* Yo = Y + (size_t)e * RPE * N;

  const int tid  = threadIdx.x;            // 0..127
  const int lane = tid & 31;
  const int wid  = tid >> 5;               // 0..3
  const int wrow = (wid & 1) * 64;
  const int wcol = (wid >> 1) * 64;
  const int rq   = tid >> 2;               // 0..31
  const int kq   = (tid & 3) * 8;

  auto issue = [&](int k0, int buf) {
#pragma unroll
    for (int c = 0; c < 4; ++c) {
      const int r = rq + 32 * c;
      async_ld_b128(&lA[buf][r * PITCH + kq], A  + (size_t)r * K + k0 + kq);
      async_ld_b128(&lB[buf][r * PITCH + kq], BT + (size_t)r * K + k0 + kq);
    }
  };

  const v8f vzero = {0.f, 0.f, 0.f, 0.f, 0.f, 0.f, 0.f, 0.f};
  v8f acc[4][4];
#pragma unroll
  for (int i = 0; i < 4; ++i)
#pragma unroll
    for (int j = 0; j < 4; ++j) acc[i][j] = vzero;

  issue(0, 0);
  wait_async0();
  __syncthreads();

  for (int k0 = 0; k0 < K; k0 += 32) {
    const int cur = (k0 >> 5) & 1;
    if (k0 + 32 < K) issue(k0 + 32, cur ^ 1);

    v16bf af[4], bf[4];
#pragma unroll
    for (int i = 0; i < 4; ++i) af[i] = a_frag_lds(lA[cur], wrow + 16 * i, lane);
#pragma unroll
    for (int j = 0; j < 4; ++j) bf[j] = b_frag_lds(lB[cur], wcol + 16 * j, lane);
#pragma unroll
    for (int i = 0; i < 4; ++i)
#pragma unroll
      for (int j = 0; j < 4; ++j) acc[i][j] = wmma_bf16(af[i], bf[j], acc[i][j]);

    wait_async0();
    __syncthreads();
  }

  const int row0 = blockIdx.y * 128 + wrow;
  const int col0 = blockIdx.x * 128 + wcol;
  const int n0 = lane & 15;
  const int ms = (lane >> 4) << 3;
#pragma unroll
  for (int i = 0; i < 4; ++i)
#pragma unroll
    for (int j = 0; j < 4; ++j)
#pragma unroll
      for (int tq = 0; tq < 8; ++tq) {
        const int r  = row0 + i * 16 + ms + tq;
        const int cc = col0 + j * 16 + n0;
        Yo[(size_t)r * N + cc] = acc[i][j][tq];
      }
}

// ---------------- 7) combine (weighted gather of <=2 expert rows) ------------
__global__ __launch_bounds__(128) void combine_kernel(
    const float* __restrict__ Y, const int* __restrict__ tokE,
    const int* __restrict__ tokC, const float* __restrict__ tokG,
    float* __restrict__ out) {
  const int sg = blockIdx.x;               // global token index
  const int g  = sg >> 10;                 // sg / Sc
  const int e1 = tokE[sg];
  const int e2 = tokE[GSc + sg];
  const float w1 = tokG[sg];
  const float w2 = tokG[GSc + sg];
  const int c1 = min(tokC[sg], Cc - 1);
  const int c2 = min(tokC[GSc + sg], Cc - 1);
  const float* y1 = Y + (((size_t)max(e1, 0) * Gc + g) * Cc + c1) * Md;
  const float* y2 = Y + (((size_t)max(e2, 0) * Gc + g) * Cc + c2) * Md;
  float* o = out + (size_t)sg * Md;
  for (int m = threadIdx.x; m < Md; m += 128) {
    float v = 0.f;
    if (e1 >= 0) v += w1 * y1[m];
    if (e2 >= 0) v += w2 * y2[m];
    o[m] = v;
  }
}

// ---------------- launch ----------------
extern "C" void kernel_launch(void* const* d_in, const int* in_sizes, int n_in,
                              void* d_out, int out_size, void* d_ws, size_t ws_size,
                              hipStream_t stream) {
  const float* x  = (const float*)d_in[0];
  const float* rw = (const float*)d_in[1];
  const float* w0 = (const float*)d_in[2];
  const float* w1 = (const float*)d_in[3];
  const float* wo = (const float*)d_in[4];
  float* out = (float*)d_out;

  uintptr_t p = (uintptr_t)d_ws;
  auto take = [&](size_t bytes) -> void* {
    uintptr_t r = (p + 255) & ~(uintptr_t)255;
    p = r + bytes;
    return (void*)r;
  };
  float*  logits = (float*) take((size_t)GSc * Ec * 4);
  int*    slot   = (int*)   take((size_t)Ec * Gc * Cc * 4);
  int*    tokE   = (int*)   take((size_t)2 * GSc * 4);
  int*    tokC   = (int*)   take((size_t)2 * GSc * 4);
  float*  tokG   = (float*) take((size_t)2 * GSc * 4);
  __bf16* Xb     = (__bf16*)take((size_t)Ec * RPE * Md * 2);
  __bf16* W0T    = (__bf16*)take((size_t)Ec * HP * Md * 2);
  __bf16* W1T    = (__bf16*)take((size_t)Ec * HP * Md * 2);
  __bf16* WoT    = (__bf16*)take((size_t)Ec * Md * HP * 2);
  __bf16* Hid    = (__bf16*)take((size_t)Ec * RPE * HP * 2);
  float*  Yx     = (float*) take((size_t)Ec * RPE * Md * 4);
  (void)ws_size; (void)in_sizes; (void)n_in; (void)out_size;

  router_kernel<<<GSc * Ec / 256, 256, 0, stream>>>(x, rw, logits);
  gating_kernel<<<Gc, 256, 0, stream>>>(logits, slot, tokE, tokC, tokG);
  gather_kernel<<<Ec * Gc * Cc, 128, 0, stream>>>(x, slot, Xb);
  // W0T/W1T: [e][HP][M] = w{0,1}[e][m][h]^T (zero rows h>=H)
  transpose_cvt_kernel<<<dim3(Md / 32, HP / 32, Ec), 256, 0, stream>>>(
      w0, W0T, Md, Hc, HP, Md);
  transpose_cvt_kernel<<<dim3(Md / 32, HP / 32, Ec), 256, 0, stream>>>(
      w1, W1T, Md, Hc, HP, Md);
  // WoT: [e][M][HP] = wo[e][h][m]^T (zero cols h>=H)
  transpose_cvt_kernel<<<dim3(HP / 32, Md / 32, Ec), 256, 0, stream>>>(
      wo, WoT, Hc, Md, Md, HP);
  gemm1_kernel<<<dim3(HP / 128, RPE / 128, Ec), 256, 0, stream>>>(Xb, W0T, W1T, Hid);
  gemm2_kernel<<<dim3(Md / 128, RPE / 128, Ec), 128, 0, stream>>>(Hid, WoT, Yx);
  combine_kernel<<<GSc, 128, 0, stream>>>(Yx, tokE, tokC, tokG, out);
}
